// ODE_RNN_61022895342226
// MI455X (gfx1250) — matrix operand order repaired and checked by reading, not compiled, not run on hardware
//
#include <hip/hip_runtime.h>
#include <hip/hip_bf16.h>

// ---------------------------------------------------------------------------
// ODE-RNN on gfx1250 (MI455X): persistent workgroups, bf16 WMMA, f32 carry.
//   L=128, BS=512, D_IN=128, LATENT=256, HID=256, H=512, 3H=1536
// Each block owns 16 batch rows for the entire 128-step scan.
// h_cat = [ode_pre, c] is never materialized: gh reads ode_pre (K<256) and
// the carry (K>=256) directly; gate combine uses f32 buffers.
// A-fragment LDS loads are software-pipelined (one-deep double buffer).
// Outputs stored non-temporal so L2 stays reserved for the weights.
// ---------------------------------------------------------------------------

typedef __attribute__((ext_vector_type(16))) __bf16 v16bf;
typedef __attribute__((ext_vector_type(8)))  float  v8f;

#define L_SEQ  128
#define BSZ    512
#define DIN    128
#define LAT    256
#define HH     512       // H = 2*LATENT
#define DROW   129       // D_IN + 1 (time channel)

// LDS row strides in elements (bf16 strides multiple of 8 for b128 alignment,
// +4-dword padding to spread rows across the 64 LDS banks)
#define SXS  136         // x tile          16 x 128 bf16
#define SCS  264         // 256-wide bf16 rows (carry / tanh-mid / ode_pre)
#define SCF  260         // 256-wide f32 rows  (carry / ode_pre, exact)

// ---- WMMA bf16 16x16x32, f32 accumulate -----------------------------------
__device__ __forceinline__ v8f wmma_bf(v16bf a, v16bf b, v8f c) {
  return __builtin_amdgcn_wmma_f32_16x16x32_bf16(
      /*neg_a=*/false, a, /*neg_b=*/false, b,
      /*c_mod=*/(short)0, c, /*reuse_a=*/false, /*reuse_b=*/false);
}

// A fragment (16x32 bf16, row-major source, row = M):
// ISA layout: lane m (0..15 / 16..31), VGPR v0..3 -> K = 2v + hi*8,
// v4..7 -> K = 16 + 2(v-4) + hi*8  => dwords {0,1,2,3,8,9,10,11} + hi*4
__device__ __forceinline__ v16bf loadA(const __bf16* base, int strideE, int lane) {
  const int r = lane & 15, hi = lane >> 4;
  const unsigned int* p = (const unsigned int*)(base + r * strideE) + hi * 4;
  union { v16bf v; uint4 q[2]; } u;
  u.q[0] = *(const uint4*)(p);
  u.q[1] = *(const uint4*)(p + 8);
  return u.v;
}

// B fragment (32x16 bf16). We compute act @ W^T, so B[k,n] = W[n,k]:
// lane's column n reads weight row n, same K order as A:
// dwords {0,1,2,3,8,9,10,11} + hi*4 of the row.
__device__ __forceinline__ v16bf loadB(const __bf16* wrow0, int strideE, int lane) {
  const int r = lane & 15, hi = lane >> 4;
  const unsigned int* p = (const unsigned int*)(wrow0 + r * strideE) + hi * 4;
  union { v16bf v; uint4 q[2]; } u;
  u.q[0] = *(const uint4*)(p);
  u.q[1] = *(const uint4*)(p + 8);
  return u.v;
}

// Branch-free transcendentals (v_exp + v_rcp, no exec-mask slow path).
__device__ __forceinline__ float fast_sigmoid(float x) {
  return __fdividef(1.0f, 1.0f + __expf(-x));
}
__device__ __forceinline__ float fast_tanh(float x) {
  return 1.0f - __fdividef(2.0f, __expf(2.0f * x) + 1.0f);
}

// ---- GRU phase: gate-column tiles held in VGPR accumulators ---------------
// NT tiles per wave. Normal steps: NT=2 covers columns [0,256) (the carry);
// last step: NT=4 covers all [0,512) so the full final hidden is produced.
template <int NT, bool LAST>
__device__ __forceinline__ void gru_phase(
    const __bf16* sX, const __bf16* sPre, const float* sPreF,
    const __bf16* __restrict__ wih, const __bf16* __restrict__ whh,
    const float* __restrict__ bih, const float* __restrict__ bhh,
    float* sCf, __bf16* sCb,
    float* __restrict__ out1, float* __restrict__ out2,
    int rowbase, int t, int wave, int lane)
{
  const int n = lane & 15, hi = lane >> 4;
  v8f Cr[NT], Cz[NT], Cin[NT], Chn[NT];
  int j0[NT];
#pragma unroll
  for (int it = 0; it < NT; ++it) {
    j0[it] = (wave * NT + it) * 16;
    const int j = j0[it] + n;
    const float br = bih[j] + bhh[j];
    const float bz = bih[512 + j] + bhh[512 + j];
    const float bi = bih[1024 + j];
    const float bh = bhh[1024 + j];
#pragma unroll
    for (int i = 0; i < 8; ++i) { Cr[it][i]=br; Cz[it][i]=bz; Cin[it][i]=bi; Chn[it][i]=bh; }
  }

  // gi = x @ w_ih^T  (K = 128), one-deep pipelined A loads
  {
    v16bf a = loadA(sX, SXS, lane);
#pragma unroll
    for (int k0 = 0; k0 < DIN; k0 += 32) {
      v16bf an = a;
      if (k0 + 32 < DIN) an = loadA(sX + k0 + 32, SXS, lane);
#pragma unroll
      for (int it = 0; it < NT; ++it) {
        Cr[it]  = wmma_bf(a, loadB(wih + (size_t)(j0[it])        * DIN + k0, DIN, lane), Cr[it]);
        Cz[it]  = wmma_bf(a, loadB(wih + (size_t)(512  + j0[it]) * DIN + k0, DIN, lane), Cz[it]);
        Cin[it] = wmma_bf(a, loadB(wih + (size_t)(1024 + j0[it]) * DIN + k0, DIN, lane), Cin[it]);
      }
      a = an;
    }
  }
  // gh = h_cat @ w_hh^T  (K = 512): A from ode_pre (K<256) / carry (K>=256)
  {
    v16bf a = loadA(sPre, SCS, lane);
#pragma unroll
    for (int k0 = 0; k0 < HH; k0 += 32) {
      v16bf an = a;
      if (k0 + 32 < HH) {
        const int kn = k0 + 32;
        const __bf16* nb = (kn < LAT) ? (sPre + kn) : (sCb + (kn - LAT));
        an = loadA(nb, SCS, lane);
      }
#pragma unroll
      for (int it = 0; it < NT; ++it) {
        Cr[it]  = wmma_bf(a, loadB(whh + (size_t)(j0[it])        * HH + k0, HH, lane), Cr[it]);
        Cz[it]  = wmma_bf(a, loadB(whh + (size_t)(512  + j0[it]) * HH + k0, HH, lane), Cz[it]);
        Chn[it] = wmma_bf(a, loadB(whh + (size_t)(1024 + j0[it]) * HH + k0, HH, lane), Chn[it]);
      }
      a = an;
    }
  }

  // h_cat values for the gate combine, read BEFORE the carry is overwritten
  float hc[NT][8];
#pragma unroll
  for (int it = 0; it < NT; ++it) {
    const int j = j0[it] + n;
#pragma unroll
    for (int i = 0; i < 8; ++i) {
      const int m = hi * 8 + i;
      hc[it][i] = (j0[it] < LAT) ? sPreF[m * SCF + j]
                                 : sCf [m * SCF + (j - LAT)];
    }
  }
  // All LDS reads of sPre/sCb/sCf (incl. other waves' gh A-fragments) must
  // complete before anyone writes the new carry.
  __syncthreads();

#pragma unroll
  for (int it = 0; it < NT; ++it) {
    const int j = j0[it] + n;
#pragma unroll
    for (int i = 0; i < 8; ++i) {
      const int m = hi * 8 + i;
      const float r  = fast_sigmoid(Cr[it][i]);
      const float z  = fast_sigmoid(Cz[it][i]);
      const float nn = fast_tanh(Cin[it][i] + r * Chn[it][i]);
      const float nh = (1.0f - z) * nn + z * hc[it][i];
      if (j0[it] < LAT) {                       // carry + per-step output
        __builtin_nontemporal_store(nh, &out1[((size_t)t * BSZ + rowbase + m) * LAT + j]);
        sCf[m * SCF + j] = nh;
        sCb[m * SCS + j] = (__bf16)nh;
      }
      if constexpr (LAST)                       // final hidden (all 512 cols)
        __builtin_nontemporal_store(nh, &out2[(size_t)(rowbase + m) * HH + j]);
    }
  }
}

// ---- main persistent kernel: 32 blocks x 256 threads (8 waves) ------------
__global__ __launch_bounds__(256)
void ode_rnn_scan_kernel(const float* __restrict__ data,
                         const __bf16* __restrict__ w1,
                         const __bf16* __restrict__ w2,
                         const __bf16* __restrict__ wih,
                         const __bf16* __restrict__ whh,
                         const float* __restrict__ b1,
                         const float* __restrict__ b2,
                         const float* __restrict__ bih,
                         const float* __restrict__ bhh,
                         float* __restrict__ out1,
                         float* __restrict__ out2)
{
  __shared__ __align__(16) __bf16 sX  [16 * SXS];  // x tile (bf16)
  __shared__ __align__(16) __bf16 sCb [16 * SCS];  // carry c (bf16, GEMM A)
  __shared__ __align__(16) __bf16 sT1 [16 * SCS];  // tanh() mid activation
  __shared__ __align__(16) __bf16 sPre[16 * SCS];  // ode_pre (bf16, GEMM A)
  __shared__ __align__(16) float  sCf [16 * SCF];  // carry c (f32, exact)
  __shared__ __align__(16) float  sPreF[16 * SCF]; // ode_pre (f32, exact)

  const int tid = threadIdx.x, lane = tid & 31, wave = tid >> 5;
  const int rowbase = blockIdx.x * 16;

  for (int idx = tid; idx < 16 * SCS / 2; idx += 256) ((unsigned int*)sCb)[idx] = 0u;
  for (int idx = tid; idx < 16 * SCF;     idx += 256) ((unsigned int*)sCf)[idx] = 0u;
  __syncthreads();

  for (int t = 0; t < L_SEQ; ++t) {
    // stage x -> LDS, two bf16 per dword store; x is read-once -> NT loads
    const float* xrow = data + ((size_t)t * BSZ + rowbase) * DROW;
    for (int idx = tid; idx < 16 * (DIN / 2); idx += 256) {
      const int m = idx >> 6, k2 = (idx & 63) << 1;
      union { __bf16 h[2]; unsigned int u; } pk;
      pk.h[0] = (__bf16)__builtin_nontemporal_load(xrow + m * DROW + k2);
      pk.h[1] = (__bf16)__builtin_nontemporal_load(xrow + m * DROW + k2 + 1);
      *(unsigned int*)(sX + m * SXS + k2) = pk.u;
    }
    const float dt = data[((size_t)t * BSZ) * DROW + DIN];

    // Phase A: t1 = tanh(c @ w1^T + b1)   (16x256, 2 N-tiles per wave)
    {
      const int n = lane & 15, hi = lane >> 4;
      v8f C[2]; int ja[2];
#pragma unroll
      for (int it = 0; it < 2; ++it) {
        ja[it] = (wave * 2 + it) * 16;
        const float b = b1[ja[it] + n];
#pragma unroll
        for (int i = 0; i < 8; ++i) C[it][i] = b;
      }
      v16bf a = loadA(sCb, SCS, lane);
#pragma unroll
      for (int k0 = 0; k0 < LAT; k0 += 32) {
        v16bf an = a;
        if (k0 + 32 < LAT) an = loadA(sCb + k0 + 32, SCS, lane);
#pragma unroll
        for (int it = 0; it < 2; ++it)
          C[it] = wmma_bf(a, loadB(w1 + (size_t)ja[it] * LAT + k0, LAT, lane), C[it]);
        a = an;
      }
#pragma unroll
      for (int it = 0; it < 2; ++it) {
        const int j = ja[it] + n;
#pragma unroll
        for (int i = 0; i < 8; ++i)
          sT1[(hi * 8 + i) * SCS + j] = (__bf16)fast_tanh(C[it][i]);
      }
    }
    __syncthreads();

    // Phase B: ode_pre = c + (t1 @ w2^T + b2) * dt  -> sPre (bf16) + sPreF (f32)
    {
      const int n = lane & 15, hi = lane >> 4;
      v8f C[2]; int ja[2];
#pragma unroll
      for (int it = 0; it < 2; ++it) {
        ja[it] = (wave * 2 + it) * 16;
        const float b = b2[ja[it] + n];
#pragma unroll
        for (int i = 0; i < 8; ++i) C[it][i] = b;
      }
      v16bf a = loadA(sT1, SCS, lane);
#pragma unroll
      for (int k0 = 0; k0 < LAT; k0 += 32) {
        v16bf an = a;
        if (k0 + 32 < LAT) an = loadA(sT1 + k0 + 32, SCS, lane);
#pragma unroll
        for (int it = 0; it < 2; ++it)
          C[it] = wmma_bf(a, loadB(w2 + (size_t)ja[it] * LAT + k0, LAT, lane), C[it]);
        a = an;
      }
#pragma unroll
      for (int it = 0; it < 2; ++it) {
        const int j = ja[it] + n;
#pragma unroll
        for (int i = 0; i < 8; ++i) {
          const int m = hi * 8 + i;
          const float pre = sCf[m * SCF + j] + C[it][i] * dt;
          sPreF[m * SCF + j] = pre;
          sPre [m * SCS + j] = (__bf16)pre;
        }
      }
    }
    __syncthreads();

    // Phase C+D: GRU gates. Columns [0,256) normally; all 512 at final step.
    if (t == L_SEQ - 1)
      gru_phase<4, true >(sX, sPre, sPreF, wih, whh, bih, bhh, sCf, sCb,
                          out1, out2, rowbase, t, wave, lane);
    else
      gru_phase<2, false>(sX, sPre, sPreF, wih, whh, bih, bhh, sCf, sCb,
                          out1, out2, rowbase, t, wave, lane);
    __syncthreads();
  }
}

// ---- one-shot f32 -> bf16 weight conversion into workspace ----------------
#define N_W1  (256 * 256)
#define N_W2  (256 * 256)
#define N_WIH (1536 * 128)
#define N_WHH (1536 * 512)
#define N_ALL (N_W1 + N_W2 + N_WIH + N_WHH)

__global__ __launch_bounds__(256)
void convert_weights_kernel(const float* __restrict__ w1f,
                            const float* __restrict__ w2f,
                            const float* __restrict__ wihf,
                            const float* __restrict__ whhf,
                            __bf16* __restrict__ ws)
{
  const size_t i = (size_t)blockIdx.x * blockDim.x + threadIdx.x;
  if (i < N_W1)                     ws[i] = (__bf16)__builtin_nontemporal_load(w1f + i);
  else if (i < N_W1 + N_W2)         ws[i] = (__bf16)__builtin_nontemporal_load(w2f + (i - N_W1));
  else if (i < N_W1 + N_W2 + N_WIH) ws[i] = (__bf16)__builtin_nontemporal_load(wihf + (i - N_W1 - N_W2));
  else if (i < (size_t)N_ALL)       ws[i] = (__bf16)__builtin_nontemporal_load(whhf + (i - N_W1 - N_W2 - N_WIH));
}

extern "C" void kernel_launch(void* const* d_in, const int* in_sizes, int n_in,
                              void* d_out, int out_size, void* d_ws, size_t ws_size,
                              hipStream_t stream) {
  (void)in_sizes; (void)n_in; (void)out_size; (void)ws_size;
  const float* data = (const float*)d_in[0];
  const float* w1f  = (const float*)d_in[1];
  const float* b1   = (const float*)d_in[2];
  const float* w2f  = (const float*)d_in[3];
  const float* b2   = (const float*)d_in[4];
  const float* wihf = (const float*)d_in[5];
  const float* whhf = (const float*)d_in[6];
  const float* bih  = (const float*)d_in[7];
  const float* bhh  = (const float*)d_in[8];

  __bf16* ws = (__bf16*)d_ws;                       // 2.23 MB of scratch used
  const __bf16* w1  = ws;
  const __bf16* w2  = ws + N_W1;
  const __bf16* wih = ws + N_W1 + N_W2;
  const __bf16* whh = ws + N_W1 + N_W2 + N_WIH;

  float* out1 = (float*)d_out;                                  // (L,BS,LAT)
  float* out2 = out1 + (size_t)L_SEQ * BSZ * LAT;               // (1,BS,H)

  const int cvt_blocks = (N_ALL + 255) / 256;
  convert_weights_kernel<<<cvt_blocks, 256, 0, stream>>>(w1f, w2f, wihf, whhf, ws);

  ode_rnn_scan_kernel<<<BSZ / 16, 256, 0, stream>>>(
      data, w1, w2, wih, whh, b1, b2, bih, bhh, out1, out2);
}
